// MultiAttentionHeads_25666724561360
// MI455X (gfx1250) — matrix operand order repaired and checked by reading, not compiled, not run on hardware
//
#include <hip/hip_runtime.h>
#include <hip/hip_bf16.h>
#include <stdint.h>

// ---------------- problem constants ----------------
#define HN    16
#define XS    1024
#define DSZ   64
#define SEQ   2048
#define BATCH 2
#define ROWS  (BATCH*SEQ)   // 4096 flattened (b,s) rows

typedef __attribute__((ext_vector_type(16))) __bf16 v16bf;
typedef __attribute__((ext_vector_type(8)))  float  v8f;

union Frag16 { v16bf v; uint4 q[2]; unsigned short h[16]; };

__device__ __forceinline__ v8f wmma_bf16(v16bf a, v16bf b, v8f c) {
  // D = A(16x32 bf16) * B(32x16 bf16) + C(16x16 f32)
  return __builtin_amdgcn_wmma_f32_16x16x32_bf16(false, a, false, b,
                                                 (short)0, c, false, false);
}

__device__ __forceinline__ v8f vzero() {
  v8f z = {0.f,0.f,0.f,0.f,0.f,0.f,0.f,0.f};
  return z;
}

__device__ __forceinline__ unsigned short f2bf(float f) {
  unsigned int u = __float_as_uint(f);
  u += 0x7FFFu + ((u >> 16) & 1u);   // round-to-nearest-even
  return (unsigned short)(u >> 16);
}

// -------- CDNA5 async global->LDS staging (ASYNCcnt-tracked DMA) ----------
#if __has_builtin(__builtin_amdgcn_global_load_async_to_lds_b128)
#define HAVE_ASYNC 1
#else
#define HAVE_ASYNC 0
#endif

// exact pointee type per compiler diagnostic: vector_size(16) int
typedef int v4i_vs __attribute__((vector_size(16)));
typedef __attribute__((address_space(1))) v4i_vs* g_v4i_ptr;   // global
typedef __attribute__((address_space(3))) v4i_vs* l_v4i_ptr;   // LDS

__device__ __forceinline__ void copy16_async(const unsigned short* g, unsigned short* l) {
#if HAVE_ASYNC
  // generic LDS pointer: low 32 bits are the LDS byte address
  __builtin_amdgcn_global_load_async_to_lds_b128(
      (g_v4i_ptr)(uintptr_t)g,
      (l_v4i_ptr)(unsigned int)(uintptr_t)l,
      0, 0);
#else
  *(uint4*)l = *(const uint4*)g;
#endif
}

#if HAVE_ASYNC
#if __has_builtin(__builtin_amdgcn_s_wait_asynccnt)
#define WAIT_ASYNC(n) __builtin_amdgcn_s_wait_asynccnt(n)
#else
#define WAIT_ASYNC(n) asm volatile("s_wait_asynccnt %0" :: "i"(n) : "memory")
#endif
#else
#define WAIT_ASYNC(n)
#endif

// ---------------- kernel 1: fp32 -> bf16 conversions (+ weight transposes) --
// xb  : [ROWS][XS]                (same layout as x)
// wbT : [3][H][D][X]              (transposed per-head projection weights)
// wobT: [D=64][H*D=1024]          (transposed output projection)
__global__ __launch_bounds__(256) void cvt_kernel(
    const float* __restrict__ x,  const float* __restrict__ Wq,
    const float* __restrict__ Wk, const float* __restrict__ Wv,
    const float* __restrict__ Wo,
    unsigned short* __restrict__ xb, unsigned short* __restrict__ wbT,
    unsigned short* __restrict__ wobT)
{
  const int N1 = ROWS*XS;          // 4,194,304
  const int N2 = 3*HN*DSZ*XS;      // 3,145,728
  const int N3 = DSZ*XS;           // 65,536
  int idx = blockIdx.x*256 + threadIdx.x;
  if (idx < N1) { xb[idx] = f2bf(x[idx]); return; }
  idx -= N1;
  if (idx < N2) {
    const int mat = idx / (HN*DSZ*XS);
    const int r   = idx % (HN*DSZ*XS);
    const int h   = r / (DSZ*XS);
    const int r2  = r % (DSZ*XS);
    const int d   = r2 / XS;
    const int xx  = r2 % XS;
    const float* W = (mat==0) ? Wq : (mat==1) ? Wk : Wv;
    wbT[idx] = f2bf(W[((size_t)h*XS + xx)*DSZ + d]);
    return;
  }
  idx -= N2;
  if (idx < N3) {
    const int e = idx / XS, c = idx % XS;
    wobT[idx] = f2bf(Wo[(size_t)c*DSZ + e]);
  }
}

// ---------------- kernel 2: QKV projections via bf16 WMMA ------------------
// One wave computes a 16x64 tile; all 8 waves in a block share (mat, head),
// so the 64x32 weight tile per K-step is staged once in LDS (async, double-
// buffered). Q,K stored [b][h][s][d] bf16; V stored transposed [b][h][d][s].
__global__ __launch_bounds__(256) void qkv_kernel(
    const unsigned short* __restrict__ xb, const unsigned short* __restrict__ wbT,
    const float* __restrict__ bq, const float* __restrict__ bk,
    const float* __restrict__ bv,
    unsigned short* __restrict__ qbuf, unsigned short* __restrict__ kbuf,
    unsigned short* __restrict__ vtbuf)
{
  __shared__ unsigned short wlds[2][DSZ*32];     // 2 x 4KB weight tiles
  const int tid   = threadIdx.x;
  const int lane  = tid & 31;
  const int lmod  = lane & 15;
  const int lhalf = lane >> 4;
  const int w     = blockIdx.x*8 + (tid >> 5);   // 0..12287
  const int mat   = w / (HN*(ROWS/16));
  const int rem   = w % (HN*(ROWS/16));
  const int h     = rem / (ROWS/16);
  const int m0    = (rem % (ROWS/16)) * 16;

  const unsigned short* aptr  = xb + (size_t)(m0 + lmod)*XS + lhalf*16;
  const unsigned short* wbase = wbT + ((size_t)(mat*HN + h)*DSZ)*XS;

  // staging map: thread -> one 16B chunk of the 64x32 tile
  const int sd = tid >> 2, sc = tid & 3;         // row d, chunk
  // prologue: stage k0 = 0
  copy16_async(wbase + (size_t)sd*XS + sc*8, &wlds[0][sd*32 + sc*8]);

  v8f acc0 = vzero(), acc1 = vzero(), acc2 = vzero(), acc3 = vzero();

  int ib = 0;
  for (int k0 = 0; k0 < XS; k0 += 32, ib ^= 1) {
    if (k0 + 32 < XS) {
      copy16_async(wbase + (size_t)sd*XS + (k0+32) + sc*8, &wlds[ib^1][sd*32 + sc*8]);
      WAIT_ASYNC(2);   // current tile's older async ops complete
    } else {
      WAIT_ASYNC(0);
    }
    __syncthreads();

    Frag16 a;
    a.q[0] = *(const uint4*)(aptr + k0);
    a.q[1] = *(const uint4*)(aptr + k0 + 8);
    if (k0 + 32 < XS) __builtin_prefetch(aptr + k0 + 32, 0, 3);

    const unsigned short* wl = wlds[ib];
    Frag16 b;
    b.q[0] = *(const uint4*)(wl + ( 0 + lmod)*32 + lhalf*16);
    b.q[1] = *(const uint4*)(wl + ( 0 + lmod)*32 + lhalf*16 + 8);
    acc0 = wmma_bf16(a.v, b.v, acc0);
    b.q[0] = *(const uint4*)(wl + (16 + lmod)*32 + lhalf*16);
    b.q[1] = *(const uint4*)(wl + (16 + lmod)*32 + lhalf*16 + 8);
    acc1 = wmma_bf16(a.v, b.v, acc1);
    b.q[0] = *(const uint4*)(wl + (32 + lmod)*32 + lhalf*16);
    b.q[1] = *(const uint4*)(wl + (32 + lmod)*32 + lhalf*16 + 8);
    acc2 = wmma_bf16(a.v, b.v, acc2);
    b.q[0] = *(const uint4*)(wl + (48 + lmod)*32 + lhalf*16);
    b.q[1] = *(const uint4*)(wl + (48 + lmod)*32 + lhalf*16 + 8);
    acc3 = wmma_bf16(a.v, b.v, acc3);

    __syncthreads();   // everyone done reading before next overwrite
  }

  const float* bias = (mat==0) ? bq : (mat==1) ? bk : bv;
  const int b  = m0 / SEQ;
  const int s0 = m0 % SEQ;

#define QKV_STORE(ACC, NT) do {                                                \
    const int d_ = (NT)*16 + lmod;                                             \
    const float bb_ = bias[h*DSZ + d_];                                        \
    _Pragma("unroll")                                                          \
    for (int r = 0; r < 8; r++) {                                              \
      const int s_ = s0 + r + lhalf*8;                                         \
      const unsigned short val_ = f2bf((ACC)[r] + bb_);                        \
      if (mat == 0)      qbuf[((size_t)(b*HN+h)*SEQ + s_)*DSZ + d_] = val_;    \
      else if (mat == 1) kbuf[((size_t)(b*HN+h)*SEQ + s_)*DSZ + d_] = val_;    \
      else               vtbuf[((size_t)(b*HN+h)*DSZ + d_)*SEQ + s_] = val_;   \
    } } while(0)

  QKV_STORE(acc0, 0); QKV_STORE(acc1, 1); QKV_STORE(acc2, 2); QKV_STORE(acc3, 3);
#undef QKV_STORE
}

// ---------------- kernel 3: flash attention --------------------------------
// 8 waves/block, one 16-query tile each; all waves share (b,h) so K/V tiles
// (32 keys) are staged once per block into LDS via async DMA, double-buffered.
__global__ __launch_bounds__(256) void attn_kernel(
    const unsigned short* __restrict__ qbuf, const unsigned short* __restrict__ kbuf,
    const unsigned short* __restrict__ vtbuf, unsigned short* __restrict__ zb)
{
  __shared__ unsigned short klds[2][32*DSZ];    // 2 x 4KB  [t][d]
  __shared__ unsigned short vlds[2][DSZ*32];    // 2 x 4KB  [d][t]
  __shared__ unsigned short plds[8][16*32];     // per-wave P tiles

  const int tid   = threadIdx.x;
  const int lane  = tid & 31;
  const int lmod  = lane & 15;
  const int lhalf = lane >> 4;
  const int wave  = tid >> 5;
  const int w     = blockIdx.x*8 + wave;        // 0..4095
  const int bh    = w >> 7;                     // shared by whole block
  const int s0    = (w & 127) * 16;

  const unsigned short* qp = qbuf  + (size_t)bh*SEQ*DSZ;
  const unsigned short* kp = kbuf  + (size_t)bh*SEQ*DSZ;
  const unsigned short* vp = vtbuf + (size_t)bh*DSZ*SEQ;
  unsigned short* pl = &plds[wave][0];

  // staging maps: one 16B chunk per thread per tile
  const int tK = tid >> 3, cK = tid & 7;        // K: 32 rows x 8 chunks
  const int dV = tid >> 2, cV = tid & 3;        // V: 64 rows x 4 chunks

  // prologue: stage tile t0 = 0
  copy16_async(kp + (size_t)tK*DSZ + cK*8,       &klds[0][tK*DSZ + cK*8]);
  copy16_async(vp + (size_t)dV*SEQ + cV*8,       &vlds[0][dV*32 + cV*8]);

  // preload Q A-fragments (K-steps d=0..31, 32..63)
  Frag16 qa0, qa1;
  {
    const unsigned short* ap = qp + (size_t)(s0 + lmod)*DSZ + lhalf*16;
    qa0.q[0] = *(const uint4*)(ap);
    qa0.q[1] = *(const uint4*)(ap + 8);
    qa1.q[0] = *(const uint4*)(ap + 32);
    qa1.q[1] = *(const uint4*)(ap + 40);
  }

  v8f o0 = vzero(), o1 = vzero(), o2 = vzero(), o3 = vzero();
  float mrun[8], lrun[8];
#pragma unroll
  for (int r = 0; r < 8; r++) { mrun[r] = -3.0e38f; lrun[r] = 0.f; }
  const float scale = 0.125f;   // 1/sqrt(64)

  int ib = 0;
  for (int t0 = 0; t0 < SEQ; t0 += 32, ib ^= 1) {
    if (t0 + 32 < SEQ) {
      copy16_async(kp + (size_t)(t0+32+tK)*DSZ + cK*8, &klds[ib^1][tK*DSZ + cK*8]);
      copy16_async(vp + (size_t)dV*SEQ + (t0+32) + cV*8, &vlds[ib^1][dV*32 + cV*8]);
      WAIT_ASYNC(2);   // this tile's 2 async ops done; next tile's 2 in flight
    } else {
      WAIT_ASYNC(0);
    }
    __syncthreads();

    const unsigned short* kt = klds[ib];
    const unsigned short* vl = vlds[ib];

    // ---- scores: two 16x16 tiles, K-dim = 64 (two WMMAs each), from LDS ----
    v8f sc0, sc1;
    {
      const unsigned short* bp = kt + ( 0 + lmod)*DSZ + lhalf*16;
      Frag16 kf; v8f c = vzero();
      kf.q[0] = *(const uint4*)(bp);      kf.q[1] = *(const uint4*)(bp + 8);
      c = wmma_bf16(qa0.v, kf.v, c);
      kf.q[0] = *(const uint4*)(bp + 32); kf.q[1] = *(const uint4*)(bp + 40);
      c = wmma_bf16(qa1.v, kf.v, c);
      sc0 = c;
    }
    {
      const unsigned short* bp = kt + (16 + lmod)*DSZ + lhalf*16;
      Frag16 kf; v8f c = vzero();
      kf.q[0] = *(const uint4*)(bp);      kf.q[1] = *(const uint4*)(bp + 8);
      c = wmma_bf16(qa0.v, kf.v, c);
      kf.q[0] = *(const uint4*)(bp + 32); kf.q[1] = *(const uint4*)(bp + 40);
      c = wmma_bf16(qa1.v, kf.v, c);
      sc1 = c;
    }

    // ---- online softmax (fp32); row m = r + 8*lhalf across lanes of a half
    float corr[8];
#pragma unroll
    for (int r = 0; r < 8; r++) {
      float a0 = sc0[r]*scale, a1 = sc1[r]*scale;
      float mx = fmaxf(a0, a1);
      mx = fmaxf(mx, __shfl_xor(mx, 1, 32));
      mx = fmaxf(mx, __shfl_xor(mx, 2, 32));
      mx = fmaxf(mx, __shfl_xor(mx, 4, 32));
      mx = fmaxf(mx, __shfl_xor(mx, 8, 32));
      const float mnew = fmaxf(mrun[r], mx);
      corr[r] = __expf(mrun[r] - mnew);
      mrun[r] = mnew;
      const float e0 = __expf(a0 - mnew);
      const float e1 = __expf(a1 - mnew);
      float rs = e0 + e1;
      rs += __shfl_xor(rs, 1, 32);
      rs += __shfl_xor(rs, 2, 32);
      rs += __shfl_xor(rs, 4, 32);
      rs += __shfl_xor(rs, 8, 32);
      lrun[r] = lrun[r]*corr[r] + rs;
      pl[(r + lhalf*8)*32 + lmod]      = f2bf(e0);
      pl[(r + lhalf*8)*32 + 16 + lmod] = f2bf(e1);
    }
#pragma unroll
    for (int r = 0; r < 8; r++) {
      o0[r] *= corr[r]; o1[r] *= corr[r]; o2[r] *= corr[r]; o3[r] *= corr[r];
    }

    // wave-local RAW on LDS P tile: wait split DS counter (CDNA5)
    asm volatile("s_wait_dscnt 0x0" ::: "memory");
    Frag16 pa;
    const unsigned short* pp = pl + lmod*32 + lhalf*16;
    pa.q[0] = *(const uint4*)(pp);
    pa.q[1] = *(const uint4*)(pp + 8);

    // ---- O += P (16x32) * V (32x64), V tile [d][t] in LDS ----
    {
      Frag16 vf;
      const unsigned short* vb;
      vb = vl + ( 0 + lmod)*32 + lhalf*16;
      vf.q[0] = *(const uint4*)(vb); vf.q[1] = *(const uint4*)(vb + 8);
      o0 = wmma_bf16(pa.v, vf.v, o0);
      vb = vl + (16 + lmod)*32 + lhalf*16;
      vf.q[0] = *(const uint4*)(vb); vf.q[1] = *(const uint4*)(vb + 8);
      o1 = wmma_bf16(pa.v, vf.v, o1);
      vb = vl + (32 + lmod)*32 + lhalf*16;
      vf.q[0] = *(const uint4*)(vb); vf.q[1] = *(const uint4*)(vb + 8);
      o2 = wmma_bf16(pa.v, vf.v, o2);
      vb = vl + (48 + lmod)*32 + lhalf*16;
      vf.q[0] = *(const uint4*)(vb); vf.q[1] = *(const uint4*)(vb + 8);
      o3 = wmma_bf16(pa.v, vf.v, o3);
    }

    __syncthreads();  // all waves done reading this buffer before reuse
  }

  // normalize and write z in super_z layout [B][S][H*D] (bf16)
  const int b = bh >> 4, h = bh & 15;
#define ATTN_STORE(ACC, NT) do {                                               \
    _Pragma("unroll")                                                          \
    for (int r = 0; r < 8; r++) {                                              \
      const float val_ = (ACC)[r] / lrun[r];                                   \
      const int zrow_ = b*SEQ + s0 + r + lhalf*8;                              \
      zb[(size_t)zrow_*(HN*DSZ) + h*DSZ + (NT)*16 + lmod] = f2bf(val_);        \
    } } while(0)
  ATTN_STORE(o0, 0); ATTN_STORE(o1, 1); ATTN_STORE(o2, 2); ATTN_STORE(o3, 3);
#undef ATTN_STORE
}

// ---------------- kernel 4: output projection (1024 -> 64), fp32 out -------
__global__ __launch_bounds__(256) void oproj_kernel(
    const unsigned short* __restrict__ zb, const unsigned short* __restrict__ wobT,
    const float* __restrict__ bo, float* __restrict__ out)
{
  const int lane  = threadIdx.x & 31;
  const int lmod  = lane & 15;
  const int lhalf = lane >> 4;
  const int w     = blockIdx.x*8 + (threadIdx.x >> 5);   // 0..255
  const int m0    = w * 16;

  const unsigned short* aptr  = zb + (size_t)(m0 + lmod)*XS + lhalf*16;
  const unsigned short* bptr0 = wobT + (size_t)( 0 + lmod)*XS + lhalf*16;
  const unsigned short* bptr1 = wobT + (size_t)(16 + lmod)*XS + lhalf*16;
  const unsigned short* bptr2 = wobT + (size_t)(32 + lmod)*XS + lhalf*16;
  const unsigned short* bptr3 = wobT + (size_t)(48 + lmod)*XS + lhalf*16;

  v8f acc0 = vzero(), acc1 = vzero(), acc2 = vzero(), acc3 = vzero();
  for (int k0 = 0; k0 < XS; k0 += 32) {
    Frag16 a;
    a.q[0] = *(const uint4*)(aptr + k0);
    a.q[1] = *(const uint4*)(aptr + k0 + 8);
    if (k0 + 32 < XS) __builtin_prefetch(aptr + k0 + 32, 0, 3);
    Frag16 b;
    b.q[0] = *(const uint4*)(bptr0 + k0); b.q[1] = *(const uint4*)(bptr0 + k0 + 8);
    acc0 = wmma_bf16(a.v, b.v, acc0);
    b.q[0] = *(const uint4*)(bptr1 + k0); b.q[1] = *(const uint4*)(bptr1 + k0 + 8);
    acc1 = wmma_bf16(a.v, b.v, acc1);
    b.q[0] = *(const uint4*)(bptr2 + k0); b.q[1] = *(const uint4*)(bptr2 + k0 + 8);
    acc2 = wmma_bf16(a.v, b.v, acc2);
    b.q[0] = *(const uint4*)(bptr3 + k0); b.q[1] = *(const uint4*)(bptr3 + k0 + 8);
    acc3 = wmma_bf16(a.v, b.v, acc3);
  }

#define OPROJ_STORE(ACC, NT) do {                                              \
    const int e_ = (NT)*16 + lmod;                                             \
    const float bb_ = bo[e_];                                                  \
    _Pragma("unroll")                                                          \
    for (int r = 0; r < 8; r++) {                                              \
      out[(size_t)(m0 + r + lhalf*8)*DSZ + e_] = (ACC)[r] + bb_;               \
    } } while(0)
  OPROJ_STORE(acc0, 0); OPROJ_STORE(acc1, 1); OPROJ_STORE(acc2, 2); OPROJ_STORE(acc3, 3);
#undef OPROJ_STORE
}

// ---------------- host launcher --------------------------------------------
extern "C" void kernel_launch(void* const* d_in, const int* in_sizes, int n_in,
                              void* d_out, int out_size, void* d_ws, size_t ws_size,
                              hipStream_t stream) {
  const float* x  = (const float*)d_in[0];
  const float* Wq = (const float*)d_in[1];
  const float* bq = (const float*)d_in[2];
  const float* Wk = (const float*)d_in[3];
  const float* bk = (const float*)d_in[4];
  const float* Wv = (const float*)d_in[5];
  const float* bv = (const float*)d_in[6];
  const float* Wo = (const float*)d_in[7];
  const float* bo = (const float*)d_in[8];
  float* out = (float*)d_out;
  char* ws = (char*)d_ws;

  // workspace layout (bytes), total ~46.1 MiB
  unsigned short* xb   = (unsigned short*)(ws + 0);          // 8,388,608
  unsigned short* wbT  = (unsigned short*)(ws + 8388608);    // 6,291,456
  unsigned short* wobT = (unsigned short*)(ws + 14680064);   //   131,072
  unsigned short* qb   = (unsigned short*)(ws + 14811136);   // 8,388,608
  unsigned short* kb   = (unsigned short*)(ws + 23199744);   // 8,388,608
  unsigned short* vtb  = (unsigned short*)(ws + 31588352);   // 8,388,608
  unsigned short* zb   = (unsigned short*)(ws + 39976960);   // 8,388,608

  const int Ntot = ROWS*XS + 3*HN*DSZ*XS + DSZ*XS;
  cvt_kernel  <<<(Ntot + 255)/256, 256, 0, stream>>>(x, Wq, Wk, Wv, Wo, xb, wbT, wobT);
  qkv_kernel  <<<1536, 256, 0, stream>>>(xb, wbT, bq, bk, bv, qb, kb, vtb);
  attn_kernel <<< 512, 256, 0, stream>>>(qb, kb, vtb, zb);
  oproj_kernel<<<  32, 256, 0, stream>>>(zb, wobT, bo, out);

  (void)in_sizes; (void)n_in; (void)out_size; (void)ws_size;
}
